// RetroEncoder_14113262534663
// MI455X (gfx1250) — matrix operand order repaired and checked by reading, not compiled, not run on hardware
//
#include <hip/hip_runtime.h>
#include <hip/hip_bf16.h>

// ---------------------------------------------------------------------------
// Types for CDNA5 WMMA (wave32)
// ---------------------------------------------------------------------------
typedef __bf16 bf16_t;
typedef __attribute__((ext_vector_type(16))) __bf16 v16bf;
typedef __attribute__((ext_vector_type(8)))  __bf16 v8bf;
typedef __attribute__((ext_vector_type(8)))  float  v8f;

#define N_NODES 200000
#define N_EDGES 400000
#define N_GRAPHS 4096
#define DD 128

// activation codes
#define AIN_NONE 0
#define AIN_RELU 1
#define AIN_ELU  2
#define AOUT_NONE  0
#define AOUT_LRELU 1
#define AOUT_RELU  2

__device__ __forceinline__ float act_in_f(float x, int a) {
  if (a == AIN_RELU) return fmaxf(x, 0.f);
  if (a == AIN_ELU)  return x > 0.f ? x : expm1f(x);
  return x;
}
__device__ __forceinline__ float act_out_f(float x, int a) {
  if (a == AOUT_LRELU) return x > 0.f ? x : 0.01f * x;
  if (a == AOUT_RELU)  return fmaxf(x, 0.f);
  return x;
}
// order-preserving float<->uint keys for atomicMax-based segment max
__device__ __forceinline__ unsigned fkey(float f) {
  unsigned u = __float_as_uint(f);
  return (u & 0x80000000u) ? ~u : (u | 0x80000000u);
}
__device__ __forceinline__ float funkey(unsigned k) {
  unsigned u = (k & 0x80000000u) ? (k & 0x7fffffffu) : ~k;
  return __uint_as_float(u);
}

// ---------------------------------------------------------------------------
// WMMA GEMM: Y[M,Ntot] = actOut( concat(act(X0 gathered), X1 gathered) @ Wb^T + bias )
//   Wb : bf16 [Ntot, Kpad] (PyTorch Linear layout, zero-padded K, Kpad % 32 == 0)
//   Template: ACT0 = input activation on X0 (compile-time), HASX1 = concat mode.
// Block: 256 thr (8 waves). Tile BM=128 x BN=128, K-step 32 (one wmma K).
// ---------------------------------------------------------------------------
#define BM 128
#define BN 128
#define KT 32
#define ASTR 40   // LDS row stride in halfs (80B, 16B aligned, bank-rotating)

template <int ACT0, bool HASX1>
__global__ __launch_bounds__(256) void gemm_bf16_wmma(
    const float* __restrict__ X0, int K0, const int* __restrict__ idx0,
    const float* __restrict__ X1, int K1, const int* __restrict__ idx1,
    const bf16_t* __restrict__ Wb, int Kpad,
    const float* __restrict__ bias,
    float* __restrict__ Y, int M, int Ntot, int actOut)
{
  __shared__ bf16_t lA[BM * ASTR];
  __shared__ bf16_t lB[BN * ASTR];
  const int tid  = threadIdx.x;
  const int wave = tid >> 5;
  const int lane = tid & 31;
  const int mbase = blockIdx.x * BM;
  const int nbase = blockIdx.y * BN;
  const int K = K0 + K1;
  const bool k0vec = (K0 & 3) == 0;   // row base 16B-aligned for float4 path

  v8f acc[8] = {};

  // staging map: thread -> (row, 16-col chunk)
  const int lrow = tid >> 1;
  const int lcs  = (tid & 1) * 16;
  const int gm_l = mbase + lrow;
  const bool mok = gm_l < M;
  int r0 = 0, r1 = 0;
  if (mok) {
    r0 = idx0 ? idx0[gm_l] : gm_l;
    if (HASX1) r1 = idx1 ? idx1[gm_l] : gm_l;
  }
  const int gn_l = nbase + lrow;
  const bool nok = gn_l < Ntot;
  const float* __restrict__ x0row = X0 + (size_t)r0 * K0;
  const float* __restrict__ x1row = HASX1 ? (X1 + (size_t)r1 * K1) : nullptr;
  const bf16_t* __restrict__ wrow = Wb + (size_t)gn_l * Kpad;

  for (int k0 = 0; k0 < Kpad; k0 += KT) {
    const int kc = k0 + lcs;
    // prefetch next K-tile (global_prefetch_b8)
    if (k0 + KT < Kpad) {
      if (mok && kc + KT < K0) __builtin_prefetch(x0row + kc + KT, 0, 0);
      if (nok) __builtin_prefetch(wrow + kc + KT, 0, 0);
    }
    // ---- stage A (fp32 -> bf16, fused gather + concat + input activation)
    float vals[16];
    if (mok && k0vec && kc + 16 <= K0) {
      const float4* p = (const float4*)(x0row + kc);
      #pragma unroll
      for (int q = 0; q < 4; ++q) {
        float4 f = p[q];
        vals[4*q+0] = act_in_f(f.x, ACT0); vals[4*q+1] = act_in_f(f.y, ACT0);
        vals[4*q+2] = act_in_f(f.z, ACT0); vals[4*q+3] = act_in_f(f.w, ACT0);
      }
    } else if (HASX1 && mok && kc >= K0 && kc + 16 <= K) {
      const float4* p = (const float4*)(x1row + (kc - K0));
      #pragma unroll
      for (int q = 0; q < 4; ++q) {
        float4 f = p[q];
        vals[4*q+0] = f.x; vals[4*q+1] = f.y; vals[4*q+2] = f.z; vals[4*q+3] = f.w;
      }
    } else {
      #pragma unroll
      for (int c = 0; c < 16; ++c) {
        int k = kc + c;
        float v = 0.f;
        if (mok && k < K) {
          if (k < K0) v = act_in_f(x0row[k], ACT0);
          else if (HASX1) v = x1row[k - K0];
        }
        vals[c] = v;
      }
    }
    v8bf o0, o1;
    #pragma unroll
    for (int q = 0; q < 8; ++q) { o0[q] = (bf16_t)vals[q]; o1[q] = (bf16_t)vals[8 + q]; }
    *(v8bf*)(&lA[lrow * ASTR + lcs])     = o0;
    *(v8bf*)(&lA[lrow * ASTR + lcs + 8]) = o1;
    // ---- stage B (already bf16, padded; 2x16B vector copies)
    {
      v8bf w0 = {}, w1 = {};
      if (nok) {
        const v8bf* p = (const v8bf*)(wrow + kc);
        w0 = p[0]; w1 = p[1];
      }
      *(v8bf*)(&lB[lrow * ASTR + lcs])     = w0;
      *(v8bf*)(&lB[lrow * ASTR + lcs + 8]) = w1;
    }
    __syncthreads();

    // ---- fragments per ISA layout (wave32)
    const int half = lane >> 4;   // which half-wave
    const int lr   = lane & 15;
    const bf16_t* ar = &lA[(wave * 16 + lr) * ASTR];
    v8bf alo = *(const v8bf*)(ar + half * 8);        // K 0..7  | 8..15
    v8bf ahi = *(const v8bf*)(ar + 16 + half * 8);   // K 16..23| 24..31
    v16bf afrag = __builtin_shufflevector(alo, ahi, 0,1,2,3,4,5,6,7,8,9,10,11,12,13,14,15);
    v16bf bfrag[8];
    #pragma unroll
    for (int t = 0; t < 8; ++t) {
      const bf16_t* br = &lB[(t * 16 + lr) * ASTR];
      v8bf b0 = *(const v8bf*)(br + half * 8);       // K 0..7  | 8..15
      v8bf b1 = *(const v8bf*)(br + 16 + half * 8);  // K 16..23| 24..31
      bfrag[t] = __builtin_shufflevector(b0, b1, 0,1,2,3,4,5,6,7,8,9,10,11,12,13,14,15);
    }
    #pragma unroll
    for (int t = 0; t < 8; ++t)
      acc[t] = __builtin_amdgcn_wmma_f32_16x16x32_bf16(
          false, afrag, false, bfrag[t], (short)0, acc[t], false, false);
    __syncthreads();
  }

  // ---- epilogue: C layout lanes0-15:M=r / lanes16-31:M=r+8, N=lane%16
  const int lr   = lane & 15;
  const int mofs = (lane >> 4) * 8;
  #pragma unroll
  for (int t = 0; t < 8; ++t) {
    int gn = nbase + t * 16 + lr;
    float bv = bias ? bias[gn] : 0.f;
    #pragma unroll
    for (int r = 0; r < 8; ++r) {
      int gm = mbase + wave * 16 + mofs + r;
      if (gm < M && gn < Ntot)
        Y[(size_t)gm * Ntot + gn] = act_out_f(acc[t][r] + bv, actOut);
    }
  }
}

// ---------------------------------------------------------------------------
// GEMV logits: z[i] = lrelu( W[0:K0].f(S0[i0]) + W[K0:].S1[i1] + b ), atomicMax seg max
// one wave per row
// ---------------------------------------------------------------------------
__global__ __launch_bounds__(256) void gemv_logits_k(
    const float* __restrict__ S0, const int* __restrict__ idx0, int act0, int K0,
    const float* __restrict__ S1, const int* __restrict__ idx1, int K1,
    const float* __restrict__ W, const float* __restrict__ b,
    const int* __restrict__ seg, float* __restrict__ zout,
    unsigned* __restrict__ smax, int M)
{
  int wid = blockIdx.x * (blockDim.x >> 5) + (threadIdx.x >> 5);
  if (wid >= M) return;
  int lane = threadIdx.x & 31;
  int r0 = idx0 ? idx0[wid] : wid;
  float s = 0.f;
  for (int k = lane; k < K0; k += 32)
    s += W[k] * act_in_f(S0[(size_t)r0 * K0 + k], act0);
  if (S1) {
    int r1 = idx1 ? idx1[wid] : wid;
    for (int k = lane; k < K1; k += 32)
      s += W[K0 + k] * S1[(size_t)r1 * K1 + k];
  }
  #pragma unroll
  for (int off = 16; off > 0; off >>= 1) s += __shfl_down(s, off, 32);
  if (lane == 0) {
    float z = s + b[0];
    z = z > 0.f ? z : 0.01f * z;
    zout[wid] = z;
    atomicMax(&smax[seg[wid]], fkey(z));
  }
}

__global__ void seg_expsum_k(float* __restrict__ z, const int* __restrict__ seg,
                             const unsigned* __restrict__ smax, float* __restrict__ ssum, int M)
{
  int i = blockIdx.x * blockDim.x + threadIdx.x;
  if (i >= M) return;
  int sg = seg[i];
  float e = __expf(z[i] - funkey(smax[sg]));
  z[i] = e;
  atomicAdd(&ssum[sg], e);
}

// out[seg[e]] += (ze[e]/ssum[seg[e]]) * V[vidx?vidx[e]:e]   (128-wide rows)
__global__ void scatter_wadd_k(const float* __restrict__ ze, const float* __restrict__ ssum,
                               const int* __restrict__ seg, const float* __restrict__ V,
                               const int* __restrict__ vidx, float* __restrict__ out, int M)
{
  int e = blockIdx.x;
  if (e >= M) return;
  int k = threadIdx.x;
  float a = ze[e] / ssum[seg[e]];
  int vr = vidx ? vidx[e] : e;
  atomicAdd(&out[(size_t)seg[e] * DD + k], a * V[(size_t)vr * DD + k]);
}

// out[seg[i]] += mask[i]*X[i]
__global__ void segsum_rows_k(const float* __restrict__ X, const float* __restrict__ rmask,
                              const int* __restrict__ seg, float* __restrict__ out, int M)
{
  int i = blockIdx.x;
  if (i >= M) return;
  int k = threadIdx.x;
  float m = rmask ? rmask[i] : 1.f;
  atomicAdd(&out[(size_t)seg[i] * DD + k], m * X[(size_t)i * DD + k]);
}

// GRU combine (+final relu): gi,gh [M,384], h [M,128] -> out [M,128]
__global__ void gru_mix_k(const float* __restrict__ gi, const float* __restrict__ gh,
                          const float* __restrict__ h, float* __restrict__ out, long M)
{
  long i = (long)blockIdx.x * blockDim.x + threadIdx.x;
  if (i >= M * DD) return;
  long row = i >> 7;
  int k = (int)(i & 127);
  const float* a = gi + row * 384;
  const float* bb = gh + row * 384;
  float r = 1.f / (1.f + __expf(-(a[k] + bb[k])));
  float z = 1.f / (1.f + __expf(-(a[128 + k] + bb[128 + k])));
  float n = tanhf(a[256 + k] + r * bb[256 + k]);
  float o = (1.f - z) * n + z * h[i];
  out[i] = fmaxf(o, 0.f);
}

__global__ void mask_mul_k(const float* __restrict__ X, const float* __restrict__ mask,
                           float* __restrict__ out, long M)
{
  long i = (long)blockIdx.x * blockDim.x + threadIdx.x;
  if (i >= M * DD) return;
  out[i] = X[i] * mask[i >> 7];
}

__global__ void copy_cols_k(const float* __restrict__ src, float* __restrict__ dst, int off)
{
  int i = blockIdx.x * blockDim.x + threadIdx.x;
  if (i >= N_GRAPHS * DD) return;
  int g = i >> 7, k = i & 127;
  dst[(size_t)g * 768 + off + k] = src[i];
}

__global__ void fill_u32_k(unsigned* __restrict__ p, unsigned v, long n)
{
  long i = (long)blockIdx.x * blockDim.x + threadIdx.x;
  if (i < n) p[i] = v;
}

__global__ void conv_w_bf16_k(const float* __restrict__ W, bf16_t* __restrict__ Wb,
                              int K, int Kpad, long total)
{
  long i = (long)blockIdx.x * blockDim.x + threadIdx.x;
  if (i >= total) return;
  long n = i / Kpad;
  int k = (int)(i % Kpad);
  Wb[i] = (bf16_t)((k < K) ? W[n * K + k] : 0.f);
}

// out[g,0:2] = H[g] @ W2^T + b2
__global__ void lin2_head_k(const float* __restrict__ H, const float* __restrict__ W,
                            const float* __restrict__ b, float* __restrict__ out)
{
  int g = blockIdx.x;
  int lane = threadIdx.x;
  #pragma unroll
  for (int j = 0; j < 2; ++j) {
    float s = 0.f;
    for (int k = lane; k < DD; k += 32) s += H[(size_t)g * DD + k] * W[j * DD + k];
    #pragma unroll
    for (int off = 16; off > 0; off >>= 1) s += __shfl_down(s, off, 32);
    if (lane == 0) out[g * 2 + j] = s + b[j];
  }
}

// ---------------------------------------------------------------------------
// Host orchestration
// ---------------------------------------------------------------------------
static size_t s_wsOff;
template <typename T>
static T* wsAlloc(void* ws, size_t count) {
  s_wsOff = (s_wsOff + 255) & ~(size_t)255;
  T* p = (T*)((char*)ws + s_wsOff);
  s_wsOff += count * sizeof(T);
  return p;
}

extern "C" void kernel_launch(void* const* d_in, const int* in_sizes, int n_in,
                              void* d_out, int out_size, void* d_ws, size_t ws_size,
                              hipStream_t stream) {
  (void)in_sizes; (void)n_in; (void)out_size; (void)ws_size;
  s_wsOff = 0;

  const int N = N_NODES, E = N_EDGES, G = N_GRAPHS;

  // ---- data inputs (setup_inputs insertion order)
  const float* node_h   = (const float*)d_in[0];
  const float* edge_e   = (const float*)d_in[1];
  const int*   src      = (const int*)d_in[2];
  const int*   dst      = (const int*)d_in[3];
  const int*   n2g      = (const int*)d_in[4];
  const int*   e2g      = (const int*)d_in[5];
  const float* node_h_p = (const float*)d_in[6];
  const float* edge_e_p = (const float*)d_in[7];
  const int*   src_p    = (const int*)d_in[8];
  const int*   dst_p    = (const int*)d_in[9];
  const int*   n2g_p    = (const int*)d_in[10];
  const float* mask_node    = (const float*)d_in[11];  // [2,N]
  const float* mask_edge    = (const float*)d_in[12];
  const float* mask_product = (const float*)d_in[13];

  // params leaves (jax pytree order: sorted dict keys)
  auto L = [&](int i) { return (const float*)d_in[14 + i]; };
  // leaf index map:
  //  gnn=0..27, gnn_p=28..55 : ctx_tf.{W,b}=+0,+1 ; ctx_gru.{Whh,Wih,bhh,bih}=+2..+5 ;
  //    layer l base=+6+8l: gru.{Whh,Wih,bhh,bih}=+0..3, proj_edge.{W,b}=+4,5, proj_node.{W,b}=+6,7 ;
  //    proj_edge1.{W,b}=+22,23 ; proj_edge2.{W,b}=+24,25 ; proj_node.{W,b}=+26,27
  //  input_edge=56,57 input_edge_p=58,59 input_node=60,61 input_node_p=62,63
  //  lin1=64,65 lin2=66,67
  //  ro_after=68 ro_before=92 ro_c=116 ro_d=140 ro_p=164 ; pool p base=rb+8p:
  //    gru.{Whh,Wih,bhh,bih}=+0..3, logits.{W,b}=+4,5, proj.{W,b}=+6,7

  auto fillU = [&](void* p, unsigned v, long n) {
    fill_u32_k<<<(int)((n + 255) / 256), 256, 0, stream>>>((unsigned*)p, v, n);
  };
  auto zeroF = [&](float* p, long n) { fillU((void*)p, 0u, n); };

  auto convW = [&](const float* W, int Nr, int K, int Kpad) -> const bf16_t* {
    bf16_t* o = wsAlloc<bf16_t>(d_ws, (size_t)Nr * Kpad);
    long total = (long)Nr * Kpad;
    conv_w_bf16_k<<<(int)((total + 255) / 256), 256, 0, stream>>>(W, o, K, Kpad, total);
    return o;
  };

  // dispatch over the 3 used (ACT0, HASX1) combinations
  auto gemm = [&](const float* X0, int K0, const int* i0, int a0,
                  const float* X1, int K1, const int* i1,
                  const bf16_t* Wb, int Kpad, const float* bias,
                  float* Y, int M, int Nt, int actOut) {
    dim3 grid((M + BM - 1) / BM, Nt / BN);
    if (X1)
      gemm_bf16_wmma<AIN_NONE, true><<<grid, 256, 0, stream>>>(
          X0, K0, i0, X1, K1, i1, Wb, Kpad, bias, Y, M, Nt, actOut);
    else if (a0 == AIN_ELU)
      gemm_bf16_wmma<AIN_ELU, false><<<grid, 256, 0, stream>>>(
          X0, K0, i0, nullptr, 0, nullptr, Wb, Kpad, bias, Y, M, Nt, actOut);
    else
      gemm_bf16_wmma<AIN_NONE, false><<<grid, 256, 0, stream>>>(
          X0, K0, i0, nullptr, 0, nullptr, Wb, Kpad, bias, Y, M, Nt, actOut);
  };
  auto gemv = [&](const float* S0, const int* i0, int a0, int K0,
                  const float* S1, const int* i1, int K1,
                  const float* W, const float* b, const int* seg,
                  float* z, unsigned* smax, int M) {
    gemv_logits_k<<<(M + 7) / 8, 256, 0, stream>>>(S0, i0, a0, K0, S1, i1, K1, W, b,
                                                   seg, z, smax, M);
  };

  // ---- activation workspace (bump-allocated; deterministic layout per call)
  float* nb0   = wsAlloc<float>(d_ws, (size_t)N * DD);
  float* nb1   = wsAlloc<float>(d_ws, (size_t)N * DD);
  float* nb2   = wsAlloc<float>(d_ws, (size_t)N * DD);
  float* nbig0 = wsAlloc<float>(d_ws, (size_t)N * 384);
  float* nbig1 = wsAlloc<float>(d_ws, (size_t)N * 384);
  float* eb0   = wsAlloc<float>(d_ws, (size_t)E * DD);
  float* eb1   = wsAlloc<float>(d_ws, (size_t)E * DD);
  float* ez    = wsAlloc<float>(d_ws, (size_t)E);        // logits / exp (reused for nodes)
  unsigned* smax = wsAlloc<unsigned>(d_ws, (size_t)N);
  float* ssum  = wsAlloc<float>(d_ws, (size_t)N);
  float* hA    = wsAlloc<float>(d_ws, (size_t)N * DD);   // final atom (branch 1)
  float* hP    = wsAlloc<float>(d_ws, (size_t)N * DD);   // final atom_p
  float* og[5];
  for (int i = 0; i < 5; ++i) og[i] = wsAlloc<float>(d_ws, (size_t)G * DD);
  float* gbond = wsAlloc<float>(d_ws, (size_t)G * DD);
  float* gctx  = wsAlloc<float>(d_ws, (size_t)G * DD);
  float* ggi   = wsAlloc<float>(d_ws, (size_t)G * 384);
  float* ggh   = wsAlloc<float>(d_ws, (size_t)G * 384);
  unsigned* gsmax = wsAlloc<unsigned>(d_ws, (size_t)G);
  float* gssum = wsAlloc<float>(d_ws, (size_t)G);
  float* gfeat = wsAlloc<float>(d_ws, (size_t)G * 768);
  float* hmid  = wsAlloc<float>(d_ws, (size_t)G * DD);

  // ---- convert all GEMM weights to bf16 (K zero-padded to mult of 32)
  struct GW {
    const bf16_t *ctx_tf, *cg_whh, *cg_wih, *pe1, *pn;
    const bf16_t *l_pn[2], *l_whh[2], *l_wih[2];
  };
  auto convGnn = [&](int gb) {
    GW w;
    w.ctx_tf = convW(L(gb + 0), 128, 128, 128);
    w.cg_whh = convW(L(gb + 2), 384, 128, 128);
    w.cg_wih = convW(L(gb + 3), 384, 128, 128);
    for (int l = 0; l < 2; ++l) {
      int base = gb + 6 + 8 * l;
      w.l_whh[l] = convW(L(base + 0), 384, 128, 128);
      w.l_wih[l] = convW(L(base + 1), 384, 128, 128);
      w.l_pn[l]  = convW(L(base + 6), 128, 128, 128);
    }
    w.pe1 = convW(L(gb + 22), 128, 256, 256);
    w.pn  = convW(L(gb + 26), 128, 128, 128);
    return w;
  };
  struct RW { const bf16_t *proj[3], *whh[3], *wih[3]; };
  auto convRo = [&](int rb) {
    RW r;
    for (int p = 0; p < 3; ++p) {
      int base = rb + 8 * p;
      r.whh[p]  = convW(L(base + 0), 384, 128, 128);
      r.wih[p]  = convW(L(base + 1), 384, 128, 128);
      r.proj[p] = convW(L(base + 6), 128, 128, 128);
    }
    return r;
  };
  const bf16_t* WnIn  = convW(L(60), 128, 74, 96);
  const bf16_t* WeIn  = convW(L(56), 128, 13, 32);
  const bf16_t* WnInP = convW(L(62), 128, 74, 96);
  const bf16_t* WeInP = convW(L(58), 128, 13, 32);
  GW g1 = convGnn(0), g2 = convGnn(28);
  RW r_before = convRo(92), r_after = convRo(68), r_c = convRo(116),
     r_d = convRo(140), r_p = convRo(164);
  const bf16_t* Wlin1 = convW(L(64), 128, 768, 768);

  // ---- one AttentiveFP branch: input proj -> GetContext -> 2 GNN layers
  auto branch = [&](const float* nodeF, const float* edgeF,
                    const bf16_t* WnI, const float* bnI,
                    const bf16_t* WeI, const float* beI,
                    int gb, const GW& w, const int* srcI, const int* dstI,
                    float* hOut, bool doBond) {
    // atom = Linear(node_h); bond = Linear(edge_e)
    gemm(nodeF, 74, nullptr, 0, nullptr, 0, nullptr, WnI, 96, bnI, nb0, N, DD, AOUT_NONE);
    gemm(edgeF, 13, nullptr, 0, nullptr, 0, nullptr, WeI, 32, beI, eb0, E, DD, AOUT_NONE);
    if (doBond) {  // bond_output = segsum(mask_edge * bond, e2g)
      zeroF(gbond, (long)G * DD);
      segsum_rows_k<<<E, DD, 0, stream>>>(eb0, mask_edge, e2g, gbond, E);
    }
    // hv_new = lrelu(proj_node(atom))
    gemm(nb0, DD, nullptr, 0, nullptr, 0, nullptr, w.pn, 128, L(gb + 27), nb1, N, DD, AOUT_LRELU);
    // he1 = lrelu(proj_edge1(concat(atom[src], bond)))
    gemm(nb0, DD, srcI, 0, eb0, DD, nullptr, w.pe1, 256, L(gb + 23), eb1, E, DD, AOUT_LRELU);
    // logits = lrelu(proj_edge2(concat(hv_new[dst], he1))) ; segment softmax over dst
    fillU(smax, 0u, N); zeroF(ssum, N);
    gemv(nb1, dstI, AIN_NONE, DD, eb1, nullptr, DD, L(gb + 24), L(gb + 25), dstI, ez, smax, E);
    seg_expsum_k<<<(E + 255) / 256, 256, 0, stream>>>(ez, dstI, smax, ssum, E);
    // he1_tf = ctx_edge_tf(he1)   (reuse eb0)
    gemm(eb1, DD, nullptr, 0, nullptr, 0, nullptr, w.ctx_tf, 128, L(gb + 1), eb0, E, DD, AOUT_NONE);
    // ctx = segsum(a * he1_tf, dst)  (ELU fused into next GEMM load)
    zeroF(nb2, (long)N * DD);
    scatter_wadd_k<<<E, DD, 0, stream>>>(ez, ssum, dstI, eb0, nullptr, nb2, E);
    // h = relu(GRU(elu(ctx), hv_new))
    gemm(nb2, DD, nullptr, AIN_ELU, nullptr, 0, nullptr, w.cg_wih, 128, L(gb + 5), nbig0, N, 384, AOUT_NONE);
    gemm(nb1, DD, nullptr, 0,       nullptr, 0, nullptr, w.cg_whh, 128, L(gb + 4), nbig1, N, 384, AOUT_NONE);
    gru_mix_k<<<(int)(((long)N * DD + 255) / 256), 256, 0, stream>>>(nbig0, nbig1, nb1, nb0, N);
    // 2 GNN layers
    for (int l = 0; l < 2; ++l) {
      int base = gb + 6 + 8 * l;
      fillU(smax, 0u, N); zeroF(ssum, N);
      gemv(nb0, dstI, AIN_NONE, DD, nb0, srcI, DD, L(base + 4), L(base + 5), dstI, ez, smax, E);
      seg_expsum_k<<<(E + 255) / 256, 256, 0, stream>>>(ez, dstI, smax, ssum, E);
      gemm(nb0, DD, nullptr, 0, nullptr, 0, nullptr, w.l_pn[l], 128, L(base + 7), nb1, N, DD, AOUT_NONE);
      zeroF(nb2, (long)N * DD);
      scatter_wadd_k<<<E, DD, 0, stream>>>(ez, ssum, dstI, nb1, srcI, nb2, E);
      gemm(nb2, DD, nullptr, AIN_ELU, nullptr, 0, nullptr, w.l_wih[l], 128, L(base + 3), nbig0, N, 384, AOUT_NONE);
      gemm(nb0, DD, nullptr, 0,       nullptr, 0, nullptr, w.l_whh[l], 128, L(base + 2), nbig1, N, 384, AOUT_NONE);
      float* outp = (l == 1) ? hOut : nb0;
      gru_mix_k<<<(int)(((long)N * DD + 255) / 256), 256, 0, stream>>>(nbig0, nbig1, nb0, outp, N);
    }
  };

  // ---- AttentiveFP readout (3 pooling rounds)
  auto readout = [&](const float* nf, const int* seg, int rb, const RW& rw, float* gout) {
    zeroF(gout, (long)G * DD);
    segsum_rows_k<<<N, DD, 0, stream>>>(nf, nullptr, seg, gout, N);
    for (int p = 0; p < 3; ++p) {
      int base = rb + 8 * p;
      fillU(gsmax, 0u, G); zeroF(gssum, G);
      gemv(gout, seg, AIN_RELU, DD, nf, nullptr, DD, L(base + 4), L(base + 5), seg, ez, gsmax, N);
      seg_expsum_k<<<(N + 255) / 256, 256, 0, stream>>>(ez, seg, gsmax, gssum, N);
      gemm(nf, DD, nullptr, 0, nullptr, 0, nullptr, rw.proj[p], 128, L(base + 7), nb2, N, DD, AOUT_NONE);
      zeroF(gctx, (long)G * DD);
      scatter_wadd_k<<<N, DD, 0, stream>>>(ez, gssum, seg, nb2, nullptr, gctx, N);
      gemm(gctx, DD, nullptr, AIN_ELU, nullptr, 0, nullptr, rw.wih[p], 128, L(base + 3), ggi, G, 384, AOUT_NONE);
      gemm(gout, DD, nullptr, 0,       nullptr, 0, nullptr, rw.whh[p], 128, L(base + 2), ggh, G, 384, AOUT_NONE);
      gru_mix_k<<<(int)(((long)G * DD + 255) / 256), 256, 0, stream>>>(ggi, ggh, gout, gout, G);
    }
  };

  // ---- forward pass
  branch(node_h,   edge_e,   WnIn,  L(61), WeIn,  L(57), 0,  g1, src,   dst,   hA, true);
  branch(node_h_p, edge_e_p, WnInP, L(63), WeInP, L(59), 28, g2, src_p, dst_p, hP, false);

  readout(hA, n2g,   92,  r_before, og[0]);
  readout(hP, n2g_p, 68,  r_after,  og[1]);
  long nelem = (long)N * DD;
  mask_mul_k<<<(int)((nelem + 255) / 256), 256, 0, stream>>>(hA, mask_node,     nb0, N);
  readout(nb0, n2g,   116, r_c, og[2]);
  mask_mul_k<<<(int)((nelem + 255) / 256), 256, 0, stream>>>(hA, mask_node + N, nb0, N);
  readout(nb0, n2g,   140, r_d, og[3]);
  mask_mul_k<<<(int)((nelem + 255) / 256), 256, 0, stream>>>(hP, mask_product,  nb0, N);
  readout(nb0, n2g_p, 164, r_p, og[4]);

  // ---- head: concat 6x128 -> lin1(relu) -> lin2
  const float* cats[6] = {og[0], og[1], og[2], og[3], og[4], gbond};
  for (int j = 0; j < 6; ++j)
    copy_cols_k<<<(G * DD + 255) / 256, 256, 0, stream>>>(cats[j], gfeat, j * DD);
  gemm(gfeat, 768, nullptr, 0, nullptr, 0, nullptr, Wlin1, 768, L(65), hmid, G, DD, AOUT_RELU);
  lin2_head_k<<<G, 32, 0, stream>>>(hmid, L(66), L(67), (float*)d_out);
}
